// DelayNetwork_62036507623825
// MI455X (gfx1250) — matrix operand order repaired and checked by reading, not compiled, not run on hardware
//
#include <hip/hip_runtime.h>

typedef __bf16 bf16_t;
typedef __attribute__((ext_vector_type(16))) __bf16 v16bf;
typedef __attribute__((ext_vector_type(8)))  __bf16 v8bf;
typedef __attribute__((ext_vector_type(8)))  float  v8f;

#define N_IN   512
#define N_REC  2048
#define N_OUT  128
#define BATCH  64
#define TSTEPS 250
#define DMAX   4

// ---------------------------------------------------------------------------
// A-operand load: 16-bit A 16x32 tile in this code's contiguous k order (not the ISA table order):
//   lane L (row = L&15, khalf = L>>4): VGPR0-7 hold K = khalf*16 + 0..15
//   (matches packed B)  -> two contiguous 16B loads.
// ---------------------------------------------------------------------------
__device__ inline v16bf load_a(const bf16_t* __restrict__ row, int k0, int khalf) {
    union { v16bf v; v8bf h[2]; } u;
    u.h[0] = *(const v8bf*)(row + k0 + khalf * 16);
    u.h[1] = *(const v8bf*)(row + k0 + khalf * 16 + 8);
    return u.v;
}

// One B tile (32x16, pre-packed so lane reads 16 contiguous bf16) against the
// four 16-row M-tiles covering the full batch of 64.
__device__ inline void mma4(const bf16_t* __restrict__ Abase, int stride, int k0,
                            int ncol, int khalf, v16bf bmat,
                            v8f& c0, v8f& c1, v8f& c2, v8f& c3) {
    v16bf a0 = load_a(Abase + (size_t)(ncol)      * stride, k0, khalf);
    v16bf a1 = load_a(Abase + (size_t)(ncol + 16) * stride, k0, khalf);
    v16bf a2 = load_a(Abase + (size_t)(ncol + 32) * stride, k0, khalf);
    v16bf a3 = load_a(Abase + (size_t)(ncol + 48) * stride, k0, khalf);
    c0 = __builtin_amdgcn_wmma_f32_16x16x32_bf16(false, a0, false, bmat, (short)0, c0, false, false);
    c1 = __builtin_amdgcn_wmma_f32_16x16x32_bf16(false, a1, false, bmat, (short)0, c1, false, false);
    c2 = __builtin_amdgcn_wmma_f32_16x16x32_bf16(false, a2, false, bmat, (short)0, c2, false, false);
    c3 = __builtin_amdgcn_wmma_f32_16x16x32_bf16(false, a3, false, bmat, (short)0, c3, false, false);
}

// ---------------------------------------------------------------------------
// Prologue pack kernels: emit bf16 weights in per-lane WMMA B layout.
// Packed index decomposition: i = ((((..tile..)*32 + lane) * 16) + j)
// with K = kt*32 + (lane>>4)*16 + j, N = nt*16 + (lane&15).
// ---------------------------------------------------------------------------
__global__ void pack_wd_kernel(const float* __restrict__ W_r,
                               const int* __restrict__ delays,
                               bf16_t* __restrict__ Wdp) {
    unsigned i = blockIdx.x * 256u + threadIdx.x;       // 4*2048*2048 elements
    int j    = i & 15;
    int lane = (i >> 4) & 31;
    int kt   = (i >> 9) & 63;
    int nt   = (i >> 15) & 127;
    int d    = (i >> 22) & 3;
    int K = kt * 32 + (lane >> 4) * 16 + j;             // source neuron n
    int N = nt * 16 + (lane & 15);                      // target neuron m
    float v = W_r[(size_t)N * N_REC + K];
    v = (delays[(size_t)K * N_REC + N] == (d + 1)) ? v : 0.0f;
    Wdp[i] = (bf16_t)v;
}

__global__ void pack_wi_kernel(const float* __restrict__ W_i, bf16_t* __restrict__ WiTp) {
    unsigned i = blockIdx.x * 256u + threadIdx.x;       // 512*2048 elements
    int j    = i & 15;
    int lane = (i >> 4) & 31;
    int kt   = (i >> 9) & 15;
    int nt   = (i >> 13) & 127;
    int K = kt * 32 + (lane >> 4) * 16 + j;
    int N = nt * 16 + (lane & 15);
    WiTp[i] = (bf16_t)W_i[(size_t)N * N_IN + K];
}

__global__ void pack_wo_kernel(const float* __restrict__ W_o, bf16_t* __restrict__ WoTp) {
    unsigned i = blockIdx.x * 256u + threadIdx.x;       // 2048*128 elements
    int j    = i & 15;
    int lane = (i >> 4) & 31;
    int kt   = (i >> 9) & 63;
    int nt   = (i >> 15) & 7;
    int K = kt * 32 + (lane >> 4) * 16 + j;
    int N = nt * 16 + (lane & 15);
    WoTp[i] = (bf16_t)W_o[(size_t)N * N_REC + K];
}

__global__ void pack_x_kernel(const float* __restrict__ x, bf16_t* __restrict__ Xbf) {
    unsigned i = blockIdx.x * 256u + threadIdx.x;       // 250*64*512 elements
    int n = i & 511;
    int b = (i >> 9) & 63;
    int t = i >> 15;
    Xbf[i] = (bf16_t)x[((size_t)b * N_IN + n) * TSTEPS + t];
}

// ---------------------------------------------------------------------------
// Per-timestep kernel. Grid = 136 blocks of 32 threads (1 wave each).
//   blocks 0..127 : recurrent tile (16 output neurons x full batch 64)
//   blocks 128..135: output layer update for step t-1 (pipelined)
// ---------------------------------------------------------------------------
__global__ __launch_bounds__(32) void step_kernel(
    int t,
    const float* __restrict__ Br, const float* __restrict__ Bo,
    const float* __restrict__ tau_rec, const float* __restrict__ tau_out,
    const float* __restrict__ thr_rec,
    const bf16_t* __restrict__ Wdp, const bf16_t* __restrict__ WiTp,
    const bf16_t* __restrict__ WoTp, const bf16_t* __restrict__ Xbf,
    bf16_t* __restrict__ ring, float* __restrict__ mem_rec,
    float* __restrict__ mem_out, float* __restrict__ zs) {

    const int lane  = threadIdx.x & 31;
    const int ncol  = lane & 15;
    const int khalf = lane >> 4;
    v8f c0 = {}, c1 = {}, c2 = {}, c3 = {};

    if (blockIdx.x < 128) {
        // ---------------- recurrent path: cur = x@WiT + sum_d z[t-d]@Wd ----
        const int nt   = blockIdx.x;
        const int col0 = nt * 16;

        {   // input contribution (K = 512)
            const bf16_t* Xt = Xbf + (size_t)t * (BATCH * N_IN);
            const bf16_t* Bp = WiTp + (size_t)nt * 16 * 512;
            for (int kt = 0; kt < 16; ++kt) {
                v16bf bmat = *(const v16bf*)(Bp + (size_t)kt * 512 + lane * 16);
                mma4(Xt, N_IN, kt * 32, ncol, khalf, bmat, c0, c1, c2, c3);
            }
        }
        // delayed recurrent contributions (only if t > d, matching reference)
        for (int d = 1; d <= DMAX; ++d) {
            if (t > d) {
                const bf16_t* Z  = ring + (size_t)((t - d) & 7) * (BATCH * N_REC);
                const bf16_t* Bp = Wdp + ((size_t)(d - 1) * 128 + nt) * (64 * 512);
                for (int kt = 0; kt < 64; ++kt) {
                    v16bf bmat = *(const v16bf*)(Bp + (size_t)kt * 512 + lane * 16);
                    mma4(Z, N_REC, kt * 32, ncol, khalf, bmat, c0, c1, c2, c3);
                }
            }
        }
        // ---------------- elementwise LIF update ---------------------------
        const float rc = expf(-0.01f / tau_rec[0]);
        const float th = thr_rec[0];
        bf16_t* zslot = ring + (size_t)(t & 7) * (BATCH * N_REC);
        const v8f cc[4] = {c0, c1, c2, c3};
        #pragma unroll
        for (int mt = 0; mt < 4; ++mt) {
            #pragma unroll
            for (int r = 0; r < 8; ++r) {
                int b = mt * 16 + khalf * 8 + r;      // C layout: M = r + (lane>=16)*8
                int n = col0 + ncol;
                size_t mi = (size_t)b * N_REC + n;
                float mem = mem_rec[mi] * rc + cc[mt][r] + Br[n];
                float z = (mem > th) ? 1.0f : 0.0f;
                mem -= z * th;
                mem_rec[mi] = mem;
                zslot[mi] = (bf16_t)z;
                zs[mi * TSTEPS + t] = z;              // (B, N_REC, T) layout
            }
        }
    } else {
        // ---------------- output layer: mem_out update with z[t-1] ---------
        if (t < 1) return;
        const int nt   = blockIdx.x - 128;            // 0..7
        const int col0 = nt * 16;
        const bf16_t* Z  = ring + (size_t)((t - 1) & 7) * (BATCH * N_REC);
        const bf16_t* Bp = WoTp + (size_t)nt * (64 * 512);
        for (int kt = 0; kt < 64; ++kt) {
            v16bf bmat = *(const v16bf*)(Bp + (size_t)kt * 512 + lane * 16);
            mma4(Z, N_REC, kt * 32, ncol, khalf, bmat, c0, c1, c2, c3);
        }
        const float oc = expf(-0.01f / tau_out[0]);
        const v8f cc[4] = {c0, c1, c2, c3};
        #pragma unroll
        for (int mt = 0; mt < 4; ++mt) {
            #pragma unroll
            for (int r = 0; r < 8; ++r) {
                int b = mt * 16 + khalf * 8 + r;
                int o = col0 + ncol;
                size_t idx = (size_t)b * N_OUT + o;
                mem_out[idx] = mem_out[idx] * oc + cc[mt][r] + Bo[o];
            }
        }
    }
}

// Final mem_out update (with z[T-1]) and write to d_out. Grid = 8 x 32.
__global__ __launch_bounds__(32) void final_out_kernel(
    const float* __restrict__ tau_out, const float* __restrict__ Bo,
    const bf16_t* __restrict__ WoTp, const bf16_t* __restrict__ ring,
    const float* __restrict__ mem_out, float* __restrict__ out) {
    const int lane  = threadIdx.x & 31;
    const int ncol  = lane & 15;
    const int khalf = lane >> 4;
    const int nt    = blockIdx.x;
    const int col0  = nt * 16;
    v8f c0 = {}, c1 = {}, c2 = {}, c3 = {};
    const bf16_t* Z  = ring + (size_t)((TSTEPS - 1) & 7) * (BATCH * N_REC);
    const bf16_t* Bp = WoTp + (size_t)nt * (64 * 512);
    for (int kt = 0; kt < 64; ++kt) {
        v16bf bmat = *(const v16bf*)(Bp + (size_t)kt * 512 + lane * 16);
        mma4(Z, N_REC, kt * 32, ncol, khalf, bmat, c0, c1, c2, c3);
    }
    const float oc = expf(-0.01f / tau_out[0]);
    const v8f cc[4] = {c0, c1, c2, c3};
    #pragma unroll
    for (int mt = 0; mt < 4; ++mt) {
        #pragma unroll
        for (int r = 0; r < 8; ++r) {
            int b = mt * 16 + khalf * 8 + r;
            int o = col0 + ncol;
            size_t idx = (size_t)b * N_OUT + o;
            out[idx] = mem_out[idx] * oc + cc[mt][r] + Bo[o];
        }
    }
}

// ---------------------------------------------------------------------------
extern "C" void kernel_launch(void* const* d_in, const int* in_sizes, int n_in,
                              void* d_out, int out_size, void* d_ws, size_t ws_size,
                              hipStream_t stream) {
    const float* x       = (const float*)d_in[0];
    const int*   delays  = (const int*)d_in[1];
    const float* W_i     = (const float*)d_in[2];
    const float* W_r     = (const float*)d_in[3];
    const float* W_o     = (const float*)d_in[4];
    const float* B_r     = (const float*)d_in[5];
    const float* B_o     = (const float*)d_in[6];
    const float* tau_rec = (const float*)d_in[7];
    const float* tau_out = (const float*)d_in[8];
    const float* thr     = (const float*)d_in[9];
    float* out = (float*)d_out;

    char* ws = (char*)d_ws;
    size_t off = 0;
    auto carve = [&](size_t bytes) -> void* {
        void* p = ws + off;
        off += (bytes + 255) & ~(size_t)255;
        return p;
    };
    bf16_t* Wdp  = (bf16_t*)carve((size_t)4 * N_REC * N_REC * 2);   // 32 MB
    bf16_t* WiTp = (bf16_t*)carve((size_t)N_IN * N_REC * 2);        //  2 MB
    bf16_t* WoTp = (bf16_t*)carve((size_t)N_REC * N_OUT * 2);       // .5 MB
    bf16_t* Xbf  = (bf16_t*)carve((size_t)TSTEPS * BATCH * N_IN * 2); // 16 MB
    char*   state_base = ws + off;
    bf16_t* ring    = (bf16_t*)carve((size_t)8 * BATCH * N_REC * 2);
    float*  mem_rec = (float*)carve((size_t)BATCH * N_REC * 4);
    float*  mem_out = (float*)carve((size_t)BATCH * N_OUT * 4);
    size_t state_bytes = (size_t)((ws + off) - state_base);

    // zero all sequential state (capture-safe memset node)
    hipMemsetAsync(state_base, 0, state_bytes, stream);

    // prologue: pack weights (WMMA B layout) + input raster to bf16
    pack_wd_kernel<<<dim3(65536), dim3(256), 0, stream>>>(W_r, delays, Wdp);
    pack_wi_kernel<<<dim3(4096),  dim3(256), 0, stream>>>(W_i, WiTp);
    pack_wo_kernel<<<dim3(1024),  dim3(256), 0, stream>>>(W_o, WoTp);
    pack_x_kernel <<<dim3(32000), dim3(256), 0, stream>>>(x, Xbf);

    float* zs = out + (size_t)BATCH * N_OUT;   // spikes after mem_out (8192 floats)

    for (int t = 0; t < TSTEPS; ++t) {
        step_kernel<<<dim3(136), dim3(32), 0, stream>>>(
            t, B_r, B_o, tau_rec, tau_out, thr,
            Wdp, WiTp, WoTp, Xbf, ring, mem_rec, mem_out, zs);
    }
    final_out_kernel<<<dim3(8), dim3(32), 0, stream>>>(
        tau_out, B_o, WoTp, ring, mem_out, out);
}
